// HyperbolicArcFaceCriterion_84696755077167
// MI455X (gfx1250) — matrix operand order repaired and checked by reading, not compiled, not run on hardware
//
#include <hip/hip_runtime.h>
#include <hip/hip_bf16.h>

// Problem sizes (fixed by the reference)
#define BB 2048
#define DD 512
#define CC 16384
#define NCHUNK 8          // C split into 8 chunks of 2048 for the fused softmax
#define CHUNKW 2048       // 2^11
#define LDSROWB 1040      // 512*2 bytes + 16B pad -> conflict-free ds_load_b128

typedef __attribute__((ext_vector_type(16))) __bf16 v16bf;
typedef __attribute__((ext_vector_type(8)))  float  v8f;

union V16 { uint4 q[2]; v16bf v; };

// ArcFace constants (margin = 0.5, scale = 64)
#define COS_M 0.8775825618903728f
#define SIN_M 0.479425538604203f
#define TH_C  (-0.8775825618903728f)   // cos(pi - m)
#define MM_C  0.2397127693021015f      // sin(pi - m) * m

static __device__ __forceinline__ unsigned short f2bf(float f) {
    unsigned int u = __float_as_uint(f);
    u += 0x7fffu + ((u >> 16) & 1u);   // round-to-nearest-even
    return (unsigned short)(u >> 16);
}

// ---------------------------------------------------------------------------
// Kernel 1: per-row prep.
//   blocks [0, C)   : weight row -> Poincare proto (tanh map + 0.95 clip),
//                     write bf16 proto row + y_sq
//   blocks [C, C+B) : embedding row -> bf16 copy + x_sq
// ---------------------------------------------------------------------------
__global__ __launch_bounds__(256)
void hyp_arcface_prep(const float* __restrict__ w, const float* __restrict__ x,
                      unsigned short* __restrict__ pb, unsigned short* __restrict__ eb,
                      float* __restrict__ ysq, float* __restrict__ xsq) {
    const int b = blockIdx.x;
    const bool isProto = (b < CC);
    const int row = isProto ? b : (b - CC);
    const float* src = (isProto ? w : x) + (size_t)row * DD;
    unsigned short* dst = (isProto ? pb : eb) + (size_t)row * DD;

    const int t = threadIdx.x;
    float v0 = src[t], v1 = src[t + 256];
    float p = v0 * v0 + v1 * v1;
    for (int off = 16; off; off >>= 1) p += __shfl_xor(p, off, 32);

    __shared__ float swr[8];
    __shared__ float sscale;
    if ((t & 31) == 0) swr[t >> 5] = p;
    __syncthreads();
    if (t == 0) {
        float s = 0.f;
        for (int i = 0; i < 8; ++i) s += swr[i];
        float scale;
        if (isProto) {
            float nrm = sqrtf(s);
            float vn  = fmaxf(nrm, 1e-7f);
            scale = tanhf(vn) / vn;
            float pn = scale * nrm;                 // ||proto||
            if (pn > 0.95f) { scale *= 0.95f / pn; pn = 0.95f; }
            ysq[row] = pn * pn;
        } else {
            scale = 1.f;
            xsq[row] = s;
        }
        sscale = scale;
    }
    __syncthreads();
    const float sc = sscale;
    dst[t]       = f2bf(v0 * sc);
    dst[t + 256] = f2bf(v1 * sc);
}

// ---------------------------------------------------------------------------
// Kernel 2: fused bf16-WMMA GEMM + hyperbolic-ArcFace epilogue + online softmax.
// Block = 256 threads = 8 waves. Each wave owns a 32(M) x 32(N) register tile
// (2x2 WMMA tiles -> 4 accumulators, 2 loads per WMMA). Block tile: 32 rows x
// 256 cols; 8 N-iterations cover one 2048-col chunk of C. D=512 -> 16 K-steps
// of K=32, with B fragments software-pipelined one K-step ahead. Epilogue uses
// v_rcp/v_sqrt instead of IEEE div/sqrt sequences.
// ---------------------------------------------------------------------------
__global__ __launch_bounds__(256)
void hyp_arcface_gemm_fused(const unsigned short* __restrict__ eb,
                            const unsigned short* __restrict__ pb,
                            const float* __restrict__ xsq, const float* __restrict__ ysq,
                            const int* __restrict__ labels,
                            float* __restrict__ rowmax, float* __restrict__ rowsum,
                            float* __restrict__ lbl_logit) {
    __shared__ __align__(16) unsigned char smemA[32 * LDSROWB];
    __shared__ float sMax[32][8], sSum[32][8], sLbl[32][8];

    const int tid = threadIdx.x, lane = tid & 31, wave = tid >> 5;
    const int half = (lane >= 16) ? 1 : 0;
    const int rowbase = blockIdx.x * 32;
    const int chunk = blockIdx.y;

    // ---- stage A tile (32 x 512 bf16) into LDS -----------------------------
    for (int i = tid; i < 2048; i += 256) {
        int r = i >> 6, kc = i & 63;
        const uint4* g = (const uint4*)(eb + (size_t)(rowbase + r) * DD + kc * 8);
        *(uint4*)(smemA + r * LDSROWB + kc * 16) = *g;
    }
    __syncthreads();

    // ---- per-lane row-constant preload (16 rows: 2 M-tiles x 8) ------------
    // state index si = mg*8 + r  <->  block row = mg*16 + half*8 + r
    float xs[16], dxs[16];
    int   lb[16];
    #pragma unroll
    for (int mg = 0; mg < 2; ++mg)
        #pragma unroll
        for (int r = 0; r < 8; ++r) {
            int grow = rowbase + mg * 16 + half * 8 + r;
            int si = mg * 8 + r;
            xs[si]  = xsq[grow];
            dxs[si] = 1.f - fminf(xs[si], 1.f - 1e-5f);
            lb[si]  = labels[grow];
        }
    float rmax[16], rsum[16], rlbl[16];
    #pragma unroll
    for (int s = 0; s < 16; ++s) { rmax[s] = -3.4e38f; rsum[s] = 0.f; rlbl[s] = -3.4e38f; }

    const int arow0 = (lane & 15);          // LDS A row, M-tile 0
    const int arow1 = 16 + (lane & 15);     // LDS A row, M-tile 1
    const unsigned char* pbb = (const unsigned char*)pb;

    for (int ni = 0; ni < 8; ++ni) {
        const int colbase = chunk * CHUNKW + ni * 256 + wave * 32;
        const int col0 = colbase + (lane & 15);
        const int col1 = col0 + 16;
        const unsigned char* bp0 = pbb + (size_t)col0 * (DD * 2) + (half ? 32 : 0);
        const unsigned char* bp1 = pbb + (size_t)col1 * (DD * 2) + (half ? 32 : 0);
        __builtin_prefetch(bp0 + 256 * (DD * 2), 0, 1);   // next N-iteration
        __builtin_prefetch(bp1 + 256 * (DD * 2), 0, 1);

        v8f acc00 = {}, acc01 = {}, acc10 = {}, acc11 = {};

        // ---- software-pipelined K loop: B for k+1 in flight during k ------
        V16 b0, b1, nb0, nb1, a0, a1;
        b0.q[0] = *(const uint4*)(bp0);
        b0.q[1] = *(const uint4*)(bp0 + 16);
        b1.q[0] = *(const uint4*)(bp1);
        b1.q[1] = *(const uint4*)(bp1 + 16);
        #pragma unroll
        for (int kk = 0; kk < 16; ++kk) {
            if (kk < 15) {
                nb0.q[0] = *(const uint4*)(bp0 + (kk + 1) * 64);
                nb0.q[1] = *(const uint4*)(bp0 + (kk + 1) * 64 + 16);
                nb1.q[0] = *(const uint4*)(bp1 + (kk + 1) * 64);
                nb1.q[1] = *(const uint4*)(bp1 + (kk + 1) * 64 + 16);
            }
            const int aoff = kk * 64 + (half ? 32 : 0);
            a0.q[0] = *(const uint4*)(smemA + arow0 * LDSROWB + aoff);
            a0.q[1] = *(const uint4*)(smemA + arow0 * LDSROWB + aoff + 16);
            a1.q[0] = *(const uint4*)(smemA + arow1 * LDSROWB + aoff);
            a1.q[1] = *(const uint4*)(smemA + arow1 * LDSROWB + aoff + 16);
            acc00 = __builtin_amdgcn_wmma_f32_16x16x32_bf16(false, a0.v, false, b0.v, (short)0, acc00, false, false);
            acc10 = __builtin_amdgcn_wmma_f32_16x16x32_bf16(false, a1.v, false, b0.v, (short)0, acc10, false, false);
            acc01 = __builtin_amdgcn_wmma_f32_16x16x32_bf16(false, a0.v, false, b1.v, (short)0, acc01, false, false);
            acc11 = __builtin_amdgcn_wmma_f32_16x16x32_bf16(false, a1.v, false, b1.v, (short)0, acc11, false, false);
            b0 = nb0;
            b1 = nb1;
        }

        // ---- fused hyperbolic-ArcFace epilogue + online softmax -----------
        const float yq0 = ysq[col0];
        const float yq1 = ysq[col1];
        const float dy0 = 1.f - fminf(yq0, 1.f - 1e-5f);
        const float dy1 = 1.f - fminf(yq1, 1.f - 1e-5f);

        auto update = [&](int si, float xy, int col, float yq, float dy) {
            float diff = fmaxf(xs[si] + yq - 2.f * xy, 0.f);
            float den  = fmaxf(dxs[si] * dy, 1e-7f);
            float arg  = fmaf(2.f * diff, __builtin_amdgcn_rcpf(den), 1.f);
            arg = fmaxf(arg, 1.f + 1e-7f);
            // exp(-arccosh(arg)) == 1 / (arg + sqrt(arg^2 - 1))
            float z    = arg + __builtin_amdgcn_sqrtf(fmaf(arg, arg, -1.f));
            float cosv = fmaf(2.f, __builtin_amdgcn_rcpf(z), -1.f);
            cosv = fminf(fmaxf(cosv, -1.f), 1.f);
            float sinv = __builtin_amdgcn_sqrtf(fmaxf(fmaf(-cosv, cosv, 1.f), 1e-7f));
            float phi  = cosv * COS_M - sinv * SIN_M;
            phi = (cosv > TH_C) ? phi : (cosv - MM_C);
            bool  isl   = (col == lb[si]);
            float logit = (isl ? phi : cosv) * 64.f;
            float nm = fmaxf(rmax[si], logit);
            rsum[si] = rsum[si] * __expf(rmax[si] - nm) + __expf(logit - nm);
            rmax[si] = nm;
            if (isl) rlbl[si] = logit;
        };

        #pragma unroll
        for (int r = 0; r < 8; ++r) {
            update(r,     acc00[r], col0, yq0, dy0);
            update(r,     acc01[r], col1, yq1, dy1);
            update(8 + r, acc10[r], col0, yq0, dy0);
            update(8 + r, acc11[r], col1, yq1, dy1);
        }
    }

    // ---- reduce across the 16 lanes of each half-wave (16 cols) ------------
    #pragma unroll
    for (int s = 0; s < 16; ++s) {
        for (int off = 8; off; off >>= 1) {
            float om = __shfl_xor(rmax[s], off, 16);
            float os = __shfl_xor(rsum[s], off, 16);
            float ol = __shfl_xor(rlbl[s], off, 16);
            float nm = fmaxf(rmax[s], om);
            rsum[s] = rsum[s] * __expf(rmax[s] - nm) + os * __expf(om - nm);
            rmax[s] = nm;
            rlbl[s] = fmaxf(rlbl[s], ol);
        }
    }
    if ((lane & 15) == 0) {
        #pragma unroll
        for (int mg = 0; mg < 2; ++mg)
            #pragma unroll
            for (int r = 0; r < 8; ++r) {
                int row = mg * 16 + half * 8 + r;
                sMax[row][wave] = rmax[mg * 8 + r];
                sSum[row][wave] = rsum[mg * 8 + r];
                sLbl[row][wave] = rlbl[mg * 8 + r];
            }
    }
    __syncthreads();

    // ---- merge 8 waves per row, emit chunk partials ------------------------
    if (tid < 32) {
        float M = -3.4e38f;
        for (int w = 0; w < 8; ++w) M = fmaxf(M, sMax[tid][w]);
        float S = 0.f, L = -3.4e38f;
        for (int w = 0; w < 8; ++w) {
            S += sSum[tid][w] * __expf(sMax[tid][w] - M);
            L  = fmaxf(L, sLbl[tid][w]);
        }
        const int grow = rowbase + tid;
        rowmax[grow * NCHUNK + chunk] = M;
        rowsum[grow * NCHUNK + chunk] = S;
        if ((labels[grow] >> 11) == chunk)   // chunk width 2048 = 2^11
            lbl_logit[grow] = L;
    }
}

// ---------------------------------------------------------------------------
// Kernel 3: merge chunk partials (log-sum-exp), mean loss over B rows.
// ---------------------------------------------------------------------------
__global__ __launch_bounds__(256)
void hyp_arcface_finalize(const float* __restrict__ rowmax, const float* __restrict__ rowsum,
                          const float* __restrict__ lbl_logit, float* __restrict__ out) {
    __shared__ float sred[256];
    float acc = 0.f;
    for (int row = threadIdx.x; row < BB; row += 256) {
        float M = -3.4e38f;
        for (int c = 0; c < NCHUNK; ++c) M = fmaxf(M, rowmax[row * NCHUNK + c]);
        float S = 0.f;
        for (int c = 0; c < NCHUNK; ++c)
            S += rowsum[row * NCHUNK + c] * __expf(rowmax[row * NCHUNK + c] - M);
        float lse = M + __logf(S);
        acc += lse - lbl_logit[row];
    }
    sred[threadIdx.x] = acc;
    __syncthreads();
    for (int s = 128; s; s >>= 1) {
        if (threadIdx.x < s) sred[threadIdx.x] += sred[threadIdx.x + s];
        __syncthreads();
    }
    if (threadIdx.x == 0) out[0] = sred[0] / (float)BB;
}

// ---------------------------------------------------------------------------
extern "C" void kernel_launch(void* const* d_in, const int* in_sizes, int n_in,
                              void* d_out, int out_size, void* d_ws, size_t ws_size,
                              hipStream_t stream) {
    const float* embeddings = (const float*)d_in[0];   // [B, D] f32
    const int*   labels     = (const int*)d_in[1];     // [B]    i32
    const float* weight     = (const float*)d_in[2];   // [C, D] f32
    float* out = (float*)d_out;

    unsigned char* ws = (unsigned char*)d_ws;
    size_t off = 0;
    auto take = [&](size_t bytes) { size_t o = off; off += (bytes + 255) & ~(size_t)255; return o; };
    unsigned short* eb  = (unsigned short*)(ws + take((size_t)BB * DD * 2)); // bf16 embeddings
    unsigned short* pb  = (unsigned short*)(ws + take((size_t)CC * DD * 2)); // bf16 protos
    float* xsq          = (float*)(ws + take((size_t)BB * 4));
    float* ysq          = (float*)(ws + take((size_t)CC * 4));
    float* rowmax       = (float*)(ws + take((size_t)BB * NCHUNK * 4));
    float* rowsum       = (float*)(ws + take((size_t)BB * NCHUNK * 4));
    float* lbl          = (float*)(ws + take((size_t)BB * 4));

    hyp_arcface_prep<<<CC + BB, 256, 0, stream>>>(weight, embeddings, pb, eb, ysq, xsq);
    hyp_arcface_gemm_fused<<<dim3(BB / 32, NCHUNK), 256, 0, stream>>>(
        eb, pb, xsq, ysq, labels, rowmax, rowsum, lbl);
    hyp_arcface_finalize<<<1, 256, 0, stream>>>(rowmax, rowsum, lbl, out);
}
